// PseudoQuantizedLinear_1597727834502
// MI455X (gfx1250) — matrix operand order repaired and checked by reading, not compiled, not run on hardware
//
#include <hip/hip_runtime.h>

typedef _Float16 v8h  __attribute__((ext_vector_type(8)));
typedef _Float16 v16h __attribute__((ext_vector_type(16)));
typedef float    v8f  __attribute__((ext_vector_type(8)));
typedef int      v4i  __attribute__((ext_vector_type(4)));

#define TOK   4096   // B*S tokens (M)
#define KIN   4096   // IN (K)
#define NOUT  4096   // OUT (N)
#define RROT  4
#define GQ    128
#define QMAXF 15.0f

// GEMM tiling
#define BM   128     // block tile M
#define BN   256     // block tile N
#define BK   64      // K per LDS stage (2 wmma K-steps)
#define BKP  72      // padded row length in halves (144B, 16B-aligned stride)

// --------------------------------------------------------------------------
// gfx1250 async global->LDS path (guarded; falls back to sync load+ds_store)
// Builtin signature (from hipcc diagnostic): (int4 AS1*, int4 AS3*, Ii, Ii)
// --------------------------------------------------------------------------
#if defined(__has_builtin)
#  if __has_builtin(__builtin_amdgcn_global_load_async_to_lds_b128) && \
      __has_builtin(__builtin_amdgcn_s_wait_asynccnt)
#    define HAVE_ASYNC_LDS 1
#  else
#    define HAVE_ASYNC_LDS 0
#  endif
#else
#  define HAVE_ASYNC_LDS 0
#endif

typedef __attribute__((address_space(1))) v4i* gptr4;
typedef __attribute__((address_space(3))) v4i* lptr4;

__device__ __forceinline__ void copy16_to_lds(const void* g, void* l) {
#if HAVE_ASYNC_LDS
  __builtin_amdgcn_global_load_async_to_lds_b128((gptr4)g, (lptr4)l, 0, 0);
#else
  *(v8h*)l = *(const v8h*)g;
#endif
}

template <int N>
__device__ __forceinline__ void wait_async_lds() {
#if HAVE_ASYNC_LDS
  __builtin_amdgcn_s_wait_asynccnt(N);
#endif
}

// ---------------------------------------------------------------------------
// x (f32) -> f16
// ---------------------------------------------------------------------------
__global__ void cvt_fp16_kernel(const float* __restrict__ in,
                                _Float16* __restrict__ out, int n) {
  int i = blockIdx.x * blockDim.x + threadIdx.x;
  int stride = gridDim.x * blockDim.x;
  for (; i < n; i += stride) out[i] = (_Float16)in[i];
}

// ---------------------------------------------------------------------------
// Weight preprocessing: scale -> Givens(fwd) -> fake-quant(4b, G=128)
//                       -> Givens(inv) -> unscale -> f16
// ---------------------------------------------------------------------------
__global__ void __launch_bounds__(256)
prep_weight_kernel(const float* __restrict__ weight,
                   const float* __restrict__ cs,
                   const float* __restrict__ angles,   // (R, KIN/2)
                   const int* __restrict__ pairs,      // (R, KIN)
                   _Float16* __restrict__ wh) {        // (NOUT, KIN)
  __shared__ float row[KIN];
  __shared__ float pmin[256], pmax[256];
  __shared__ float gscale[KIN / GQ], gzp[KIN / GQ];

  const int r = blockIdx.x;
  const int t = threadIdx.x;

  for (int e = t; e < KIN; e += 256)
    row[e] = weight[(size_t)r * KIN + e] * cs[e];
  __syncthreads();

  for (int k = 0; k < RROT; ++k) {
    for (int p = t; p < KIN / 2; p += 256) {
      int i = pairs[k * KIN + 2 * p];
      int j = pairs[k * KIN + 2 * p + 1];
      float a = row[i], b = row[j];
      float ang = angles[k * (KIN / 2) + p];
      float c = cosf(ang), s = sinf(ang);
      row[i] = a * c - b * s;
      row[j] = a * s + b * c;
    }
    __syncthreads();
  }

  {
    int g = t >> 3, s8 = t & 7;
    float mn = 3.4e38f, mx = -3.4e38f;
    int base = g * GQ + s8 * 16;
    for (int e2 = 0; e2 < 16; ++e2) {
      float v = row[base + e2];
      mn = fminf(mn, v);
      mx = fmaxf(mx, v);
    }
    pmin[t] = mn; pmax[t] = mx;
    __syncthreads();
    if (s8 == 0) {
      for (int u = 1; u < 8; ++u) {
        mn = fminf(mn, pmin[t + u]);
        mx = fmaxf(mx, pmax[t + u]);
      }
      float scale = fmaxf((mx - mn) / QMAXF, 1e-5f);
      gscale[g] = scale;
      gzp[g] = rintf(-mn / scale);
    }
    __syncthreads();
  }
  for (int e = t; e < KIN; e += 256) {
    int g = e >> 7;
    float scale = gscale[g], zp = gzp[g];
    float q = rintf(row[e] / scale) + zp;
    q = fminf(fmaxf(q, 0.0f), QMAXF);
    row[e] = (q - zp) * scale;
  }
  __syncthreads();

  for (int k = RROT - 1; k >= 0; --k) {
    for (int p = t; p < KIN / 2; p += 256) {
      int i = pairs[k * KIN + 2 * p];
      int j = pairs[k * KIN + 2 * p + 1];
      float a = row[i], b = row[j];
      float ang = -angles[k * (KIN / 2) + p];
      float c = cosf(ang), s = sinf(ang);
      row[i] = a * c - b * s;
      row[j] = a * s + b * c;
    }
    __syncthreads();
  }

  for (int e = t; e < KIN; e += 256)
    wh[(size_t)r * KIN + e] = (_Float16)(row[e] / cs[e]);
}

// ---------------------------------------------------------------------------
// GEMM: out[m,n] = sum_k x[m,k] * W[n,k] + bias[n]
// Block tile 128x256, 8 waves (2x4), wave tile 64x64 = 4x4 wmma fragments.
// BK=64 per stage, double-buffered LDS filled by async global->LDS copies.
// ---------------------------------------------------------------------------
__device__ __forceinline__ v16h ld_frag16(const _Float16* p0, const _Float16* p1) {
  v8h a = *(const v8h*)p0;
  v8h b = *(const v8h*)p1;
  v16h r;
#pragma unroll
  for (int i = 0; i < 8; ++i) { r[i] = a[i]; r[i + 8] = b[i]; }
  return r;
}

__global__ void __launch_bounds__(256)
gemm_f16_wmma_kernel(const _Float16* __restrict__ A,   // TOK x KIN
                     const _Float16* __restrict__ Bw,  // NOUT x KIN
                     const float* __restrict__ bias,
                     float* __restrict__ out) {        // TOK x NOUT (f32)
  __shared__ __align__(16) _Float16 As[2][BM][BKP];
  __shared__ __align__(16) _Float16 Bs[2][BN][BKP];

  const int t    = threadIdx.x;
  const int lane = t & 31;
  const int wave = t >> 5;      // 0..7
  const int wm   = wave & 1;    // 0..1 -> 64-row slab
  const int wn   = wave >> 1;   // 0..3 -> 64-col slab
  const int m0   = blockIdx.y * BM;
  const int n0   = blockIdx.x * BN;
  const int h    = lane >> 4;   // half-wave select
  const int rl   = lane & 15;

  v8f acc[4][4];
#pragma unroll
  for (int i = 0; i < 4; ++i)
#pragma unroll
    for (int j = 0; j < 4; ++j)
#pragma unroll
      for (int v = 0; v < 8; ++v) acc[i][j][v] = 0.0f;

  // Tile loader: A = 128 rows x 8 segs, B = 256 rows x 8 segs of 16B each.
  // 256 threads -> 4 passes for A, 8 passes for B; 12 async ops per thread.
  auto load_tile = [&](int kb, int buf) {
#pragma unroll
    for (int it = 0; it < 4; ++it) {
      int idx = it * 256 + t;
      int row = idx >> 3, seg = idx & 7;
      copy16_to_lds(A + (size_t)(m0 + row) * KIN + kb + seg * 8,
                    &As[buf][row][seg * 8]);
    }
#pragma unroll
    for (int it = 0; it < 8; ++it) {
      int idx = it * 256 + t;
      int row = idx >> 3, seg = idx & 7;
      copy16_to_lds(Bw + (size_t)(n0 + row) * KIN + kb + seg * 8,
                    &Bs[buf][row][seg * 8]);
    }
  };

  const int NT = KIN / BK;      // 64 K-stages
  load_tile(0, 0);

  for (int i = 0; i < NT; ++i) {
    const int cur = i & 1;
    const bool more = (i + 1 < NT);
    if (more) {
      load_tile((i + 1) * BK, cur ^ 1);
      // our previous batch (12 ops) completes in-order before the new batch
      wait_async_lds<12>();
    } else {
      wait_async_lds<0>();
    }
    __syncthreads();

#pragma unroll
    for (int ks = 0; ks < 2; ++ks) {
      const int kofs = ks * 32;
      // A fragments: lanes 0-15 hold K0-7 & K16-23; lanes 16-31 K8-15 & K24-31
      v16h af[4];
#pragma unroll
      for (int fm = 0; fm < 4; ++fm) {
        const _Float16* ap = &As[cur][wm * 64 + fm * 16 + rl][kofs];
        af[fm] = ld_frag16(ap + h * 8, ap + 16 + h * 8);
      }
      // B fragments: lanes 0-15 hold K0-7 & K16-23; lanes 16-31 K8-15 & K24-31
#pragma unroll
      for (int fn = 0; fn < 4; ++fn) {
        const _Float16* bp = &Bs[cur][wn * 64 + fn * 16 + rl][kofs];
        v16h bf = ld_frag16(bp + h * 8, bp + 16 + h * 8);
#pragma unroll
        for (int fm = 0; fm < 4; ++fm) {
          acc[fm][fn] = __builtin_amdgcn_wmma_f32_16x16x32_f16(
              false, af[fm], false, bf, (short)0, acc[fm][fn], false, false);
        }
      }
    }
    __syncthreads();   // all waves done reading buf[cur] before it is refilled
  }

  // epilogue: C/D layout -> lane rl = column, VGPR v + h*8 = row
#pragma unroll
  for (int fn = 0; fn < 4; ++fn) {
    int n = n0 + wn * 64 + fn * 16 + rl;
    float bv = bias[n];
#pragma unroll
    for (int fm = 0; fm < 4; ++fm) {
      int mbase = m0 + wm * 64 + fm * 16 + h * 8;
#pragma unroll
      for (int v = 0; v < 8; ++v) {
        out[(size_t)(mbase + v) * NOUT + n] = acc[fm][fn][v] + bv;
      }
    }
  }
}

// ---------------------------------------------------------------------------
extern "C" void kernel_launch(void* const* d_in, const int* in_sizes, int n_in,
                              void* d_out, int out_size, void* d_ws, size_t ws_size,
                              hipStream_t stream) {
  (void)in_sizes; (void)n_in; (void)out_size; (void)ws_size;

  const float* x      = (const float*)d_in[0];
  const float* weight = (const float*)d_in[1];
  const float* cs     = (const float*)d_in[2];
  const float* ang    = (const float*)d_in[3];
  const float* bias   = (const float*)d_in[4];
  const int*   pairs  = (const int*)d_in[5];
  float* out = (float*)d_out;

  // workspace: [x as f16 : 32MB][W as f16 : 32MB]
  _Float16* xh = (_Float16*)d_ws;
  _Float16* wh = xh + (size_t)TOK * KIN;

  cvt_fp16_kernel<<<2048, 256, 0, stream>>>(x, xh, TOK * KIN);
  prep_weight_kernel<<<NOUT, 256, 0, stream>>>(weight, cs, ang, pairs, wh);
  gemm_f16_wmma_kernel<<<dim3(NOUT / BN, TOK / BM), 256, 0, stream>>>(
      xh, wh, bias, out);
}
